// TimeMix_40106404610569
// MI455X (gfx1250) — matrix operand order repaired and checked by reading, not compiled, not run on hardware
//
#include <hip/hip_runtime.h>
#include <hip/hip_bf16.h>
#include <math.h>

typedef __bf16 bf16;
typedef __attribute__((ext_vector_type(16))) __bf16 v16bf;
typedef __attribute__((ext_vector_type(8)))  float  v8f;

static constexpr int Bx = 4;
static constexpr int Tt = 2048;
static constexpr int Cc = 2048;
static constexpr int Hh = 32;
static constexpr int Kk = 64;
static constexpr int Vv = 64;

// ===========================================================================
// Fragment-swizzled tile layouts (GEMM loads fragments directly from global
// memory as 32 contiguous bytes per lane -- no LDS, no repacking).
//
// B (weights), WMMA B-matrix 32x16 bf16 per tile:
//   tile (kb, nb) of [Kd/32][N/16], 512 halves contiguous.
//   lane = hi*16 + n15 ; element kk in 0..15 ;  K = kb*32 + (kk>>3)*16 + hi*8 + (kk&7),
//   Ncol = nb*16 + n15 ; offset = (kb*NT + nb)*512 + lane*16 + kk.
//
// A (activations), WMMA A-matrix 16x32 bf16 per tile:
//   tile (mt, kt) of [M/16][Kd/32], 512 halves contiguous.
//   lane = hi*16 + m15 ; element p = 2v+e ;
//   K = kt*32 + (v>>2)*16 + hi*8 + (v&3)*2 + e ; M = mt*16 + m15 ;
//   offset = (mt*KT + kt)*512 + lane*16 + p.
// ===========================================================================

// ---- fp32 weights -> bf16, B-tile swizzle (output-index driven) -----------
__global__ void cvt_swzB_kernel(const float* __restrict__ w, bf16* __restrict__ o,
                                int N, int total) {
    size_t idx = (size_t)blockIdx.x * 256 + threadIdx.x;
    if (idx >= (size_t)total) return;
    size_t tile = idx >> 9;
    int within  = (int)(idx & 511);
    int lane = within >> 4, kk = within & 15;
    int NT = N >> 4;
    int kb = (int)(tile / NT), nb = (int)(tile % NT);
    int hi = lane >> 4, n15 = lane & 15;
    int krow = (kb << 5) + ((kk >> 3) << 4) + (hi << 3) + (kk & 7);
    int ncol = (nb << 4) + n15;
    o[idx] = (bf16)w[(size_t)krow * N + ncol];
}

// ---- time-shift mix -> bf16, A-tile swizzle -------------------------------
__global__ void mix_swz_kernel(const float* __restrict__ x, const float* __restrict__ tm,
                               bf16* __restrict__ out) {
    size_t idx = (size_t)blockIdx.x * 256 + threadIdx.x;    // over B*T*C
    size_t tile = idx >> 9;
    int within  = (int)(idx & 511);
    int lane = within >> 4, p = within & 15;
    const int KT = Cc >> 5;
    int mt = (int)(tile / KT), kt = (int)(tile % KT);
    int hi = lane >> 4, m15 = lane & 15;
    int m = (mt << 4) + m15;                                // row in [0, B*T)
    int v = p >> 1, e = p & 1;
    int c = (kt << 5) + ((v >> 2) << 4) + (hi << 3) + ((v & 3) << 1) + e;
    int t = m & (Tt - 1);
    float xv  = x[(size_t)m * Cc + c];
    float xxv = (t == 0) ? 0.f : x[(size_t)(m - 1) * Cc + c];
    float mm  = tm[c];
    out[idx] = (bf16)(xv * mm + xxv * (1.f - mm));
}

// ---------------------------------------------------------------------------
// LDS-free bf16 WMMA GEMM on swizzled operands.
//   C[M,N](f32,row-major) = A(swz)[M,Kd] * B(swz)[Kd,N]
// 256 threads = 8 waves; block tile 128x128; wave tile 32x64 = 2x4 WMMA tiles.
// Double-buffered fragment loads, branchless tail (clamped re-load of last
// tile instead of per-iteration guards) -> single straight-line loop body.
// ---------------------------------------------------------------------------
__global__ __launch_bounds__(256)
void gemm_bf16_swz_kernel(const bf16* __restrict__ Asw, const bf16* __restrict__ Bsw,
                          float* __restrict__ Cmat, int N, int Kd) {
    const int tid  = threadIdx.x;
    const int lane = tid & 31;
    const int wid  = tid >> 5;
    const int wr   = wid >> 1;                  // 0..3 (M dir)
    const int wc   = wid & 1;                   // 0..1 (N dir)
    const int hi   = lane >> 4;
    const int l15  = lane & 15;
    const int KT   = Kd >> 5;
    const int NT   = N >> 4;
    const int mt0  = (blockIdx.y << 3) + (wr << 1);   // m-tile base (of 8/block)
    const int nt0  = (blockIdx.x << 3) + (wc << 2);   // n-tile base (of 8/block)
    const size_t kStrideB = (size_t)NT << 9;          // halves per B k-slab

    const bf16* aPtr[2];
    const bf16* bPtr[4];
    #pragma unroll
    for (int mi = 0; mi < 2; ++mi)
        aPtr[mi] = Asw + ((size_t)(mt0 + mi) * KT << 9) + lane * 16;
    #pragma unroll
    for (int ni = 0; ni < 4; ++ni)
        bPtr[ni] = Bsw + ((size_t)(nt0 + ni) << 9) + lane * 16;

    v8f acc[2][4];
    #pragma unroll
    for (int mi = 0; mi < 2; ++mi)
        #pragma unroll
        for (int ni = 0; ni < 4; ++ni)
            acc[mi][ni] = (v8f){0.f,0.f,0.f,0.f,0.f,0.f,0.f,0.f};

    v16bf aF[2][2], bF[2][4];
    #pragma unroll
    for (int mi = 0; mi < 2; ++mi) aF[0][mi] = *(const v16bf*)(aPtr[mi]);
    #pragma unroll
    for (int ni = 0; ni < 4; ++ni) bF[0][ni] = *(const v16bf*)(bPtr[ni]);

    int cur = 0;
    for (int kt = 0; kt < KT; ++kt) {
        const int nxt = cur ^ 1;
        // branchless clamp: last iteration harmlessly re-loads the last tile
        const int ktn = (kt + 1 < KT) ? (kt + 1) : kt;
        const int ktp = (kt + 2 < KT) ? (kt + 2) : kt;
        #pragma unroll
        for (int mi = 0; mi < 2; ++mi)
            aF[nxt][mi] = *(const v16bf*)(aPtr[mi] + ((size_t)ktn << 9));
        #pragma unroll
        for (int ni = 0; ni < 4; ++ni)
            bF[nxt][ni] = *(const v16bf*)(bPtr[ni] + (size_t)ktn * kStrideB);
        __builtin_prefetch(aPtr[0] + ((size_t)ktp << 9), 0, 1);
        #pragma unroll
        for (int mi = 0; mi < 2; ++mi)
            #pragma unroll
            for (int ni = 0; ni < 4; ++ni)
                acc[mi][ni] = __builtin_amdgcn_wmma_f32_16x16x32_bf16(
                    false, aF[cur][mi], false, bF[cur][ni], (short)0,
                    acc[mi][ni], false, false);
        cur = nxt;
    }

    // D layout: vgpr i -> M = hi*8 + i, N = l15 (ISA 16x16 f32 C/D)
    #pragma unroll
    for (int mi = 0; mi < 2; ++mi)
        #pragma unroll
        for (int ni = 0; ni < 4; ++ni)
            #pragma unroll
            for (int i = 0; i < 8; ++i) {
                int m = ((mt0 + mi) << 4) + hi * 8 + i;
                int n = ((nt0 + ni) << 4) + l15;
                Cmat[(size_t)m * N + n] = acc[mi][ni][i];
            }
}

// ---------------------------------------------------------------------------
// RWKV-v5 sequential scan: one block per (b,h); thread j owns V-column j,
// 64-entry state column register-resident; r/k broadcast via LDS.
// ---------------------------------------------------------------------------
__global__ __launch_bounds__(64)
void scan_kernel(const float* __restrict__ Pr, const float* __restrict__ Pk,
                 const float* __restrict__ Pv, const float* __restrict__ state_in,
                 const float* __restrict__ decay, const float* __restrict__ faaaa,
                 float* __restrict__ Sout, float* __restrict__ state_out) {
    const int bh = blockIdx.x;
    const int b  = bh / Hh;
    const int h  = bh % Hh;
    const int j  = threadIdx.x;

    __shared__ float ws[Kk], us[Kk], rs[Kk], ks[Kk];
    ws[j] = __expf(-__expf(decay[h * Kk + j]));
    us[j] = faaaa[h * Kk + j];

    float S[Kk];
    #pragma unroll
    for (int k = 0; k < Kk; ++k)
        S[k] = state_in[(((size_t)b * Hh + h) * Kk + k) * Vv + j];
    __syncthreads();

    for (int t = 0; t < Tt; ++t) {
        size_t base = ((size_t)b * Tt + t) * Cc + (size_t)h * Kk;
        rs[j] = Pr[base + j];
        ks[j] = Pk[base + j];
        float vj = Pv[base + j];
        __syncthreads();
        float acc = 0.f;
        #pragma unroll
        for (int k = 0; k < Kk; ++k) {
            float kv = ks[k] * vj;
            acc  += rs[k] * fmaf(us[k], kv, S[k]);
            S[k]  = fmaf(ws[k], S[k], kv);
        }
        Sout[base + j] = acc;
        __syncthreads();
    }
    #pragma unroll
    for (int k = 0; k < Kk; ++k)
        state_out[(((size_t)b * Hh + h) * Kk + k) * Vv + j] = S[k];
}

// ---------------------------------------------------------------------------
// GroupNorm over V=64 + gamma/beta + SiLU(g) gate -> bf16 in A-tile swizzle.
// One wave per head (wave32); lane handles channels lane and lane+32.
// ---------------------------------------------------------------------------
__global__ __launch_bounds__(256)
void gnorm_swz_kernel(const float* __restrict__ Sout, const float* __restrict__ Pg,
                      const float* __restrict__ gamma, const float* __restrict__ beta,
                      bf16* __restrict__ Xo) {
    const int blk  = blockIdx.x;            // B*T*4
    const int bt   = blk >> 2;
    const int hg   = blk & 3;
    const int w    = threadIdx.x >> 5;
    const int lane = threadIdx.x & 31;
    const int h    = hg * 8 + w;
    const size_t base = (size_t)bt * Cc + (size_t)h * Vv;

    float o0 = Sout[base + lane]      * 0.125f;     // HEAD_SIZE_DIVISOR
    float o1 = Sout[base + lane + 32] * 0.125f;
    float s = o0 + o1;
    float q = o0 * o0 + o1 * o1;
    #pragma unroll
    for (int off = 16; off > 0; off >>= 1) {
        s += __shfl_xor(s, off, 32);
        q += __shfl_xor(q, off, 32);
    }
    float mu  = s * (1.f / 64.f);
    float var = q * (1.f / 64.f) - mu * mu;
    float inv = rsqrtf(var + 1e-5f);

    const int mt = bt >> 4, m15 = bt & 15;
    #pragma unroll
    for (int e = 0; e < 2; ++e) {
        int   c = h * Vv + lane + e * 32;
        float o = e ? o1 : o0;
        float y = (o - mu) * inv * gamma[c] + beta[c];
        float g = Pg[(size_t)bt * Cc + c];
        float sil = g / (1.f + __expf(-g));
        // A-tile swizzled store: row m = bt, col (K) = c
        int kt = c >> 5, r = c & 31;
        int group = (r >> 4) & 1, hi2 = (r >> 3) & 1, w2 = (r >> 1) & 3, e2 = r & 1;
        int p = (((group << 2) | w2) << 1) | e2;
        int lane_s = (hi2 << 4) + m15;
        size_t off2 = (((size_t)mt * (Cc >> 5) + kt) << 9) + lane_s * 16 + p;
        Xo[off2] = (bf16)(y * sil);
    }
}

// ---------------------------------------------------------------------------
extern "C" void kernel_launch(void* const* d_in, const int* in_sizes, int n_in,
                              void* d_out, int out_size, void* d_ws, size_t ws_size,
                              hipStream_t stream) {
    (void)in_sizes; (void)n_in; (void)out_size; (void)ws_size;
    const float* x     = (const float*)d_in[0];
    const float* state = (const float*)d_in[1];
    const float* tmk   = (const float*)d_in[2];
    const float* tmv   = (const float*)d_in[3];
    const float* tmr   = (const float*)d_in[4];
    const float* tmg   = (const float*)d_in[5];
    const float* decay = (const float*)d_in[6];
    const float* faaaa = (const float*)d_in[7];
    const float* Wr    = (const float*)d_in[8];
    const float* Wk    = (const float*)d_in[9];
    const float* Wv    = (const float*)d_in[10];
    const float* Wg    = (const float*)d_in[11];
    const float* Wo    = (const float*)d_in[12];
    const float* gamma = (const float*)d_in[13];
    const float* beta  = (const float*)d_in[14];
    float* out = (float*)d_out;

    const size_t C2  = (size_t)Cc * Cc;       // 4,194,304
    const size_t BT  = (size_t)Bx * Tt;       // 8,192
    const size_t BTC = BT * Cc;               // 16,777,216

    // workspace layout (~411 MB)
    bf16* wWr  = (bf16*)d_ws;
    bf16* wWk  = wWr + C2;
    bf16* wWv  = wWk + C2;
    bf16* wWg  = wWv + C2;
    bf16* wWo  = wWg + C2;
    bf16* Abuf = wWo + C2;                    // swizzled bf16 activations (reused)
    float* Pr  = (float*)(Abuf + BTC);
    float* Pk  = Pr + BTC;
    float* Pv  = Pk + BTC;
    float* Pg  = Pv + BTC;
    float* Sout = Pg + BTC;
    bf16* Xo   = Abuf;                        // reuse after projection GEMMs

    const int cvtGrid = (int)((C2 + 255) / 256);
    cvt_swzB_kernel<<<cvtGrid, 256, 0, stream>>>(Wr, wWr, Cc, (int)C2);
    cvt_swzB_kernel<<<cvtGrid, 256, 0, stream>>>(Wk, wWk, Cc, (int)C2);
    cvt_swzB_kernel<<<cvtGrid, 256, 0, stream>>>(Wv, wWv, Cc, (int)C2);
    cvt_swzB_kernel<<<cvtGrid, 256, 0, stream>>>(Wg, wWg, Cc, (int)C2);
    cvt_swzB_kernel<<<cvtGrid, 256, 0, stream>>>(Wo, wWo, Cc, (int)C2);

    const int mixGrid = (int)(BTC / 256);
    dim3 gemmGrid(Cc / 128, (int)(BT / 128));

    // projections: r, k, v, g  (mix -> swizzled bf16 -> WMMA GEMM -> f32)
    mix_swz_kernel<<<mixGrid, 256, 0, stream>>>(x, tmr, Abuf);
    gemm_bf16_swz_kernel<<<gemmGrid, 256, 0, stream>>>(Abuf, wWr, Pr, Cc, Cc);
    mix_swz_kernel<<<mixGrid, 256, 0, stream>>>(x, tmk, Abuf);
    gemm_bf16_swz_kernel<<<gemmGrid, 256, 0, stream>>>(Abuf, wWk, Pk, Cc, Cc);
    mix_swz_kernel<<<mixGrid, 256, 0, stream>>>(x, tmv, Abuf);
    gemm_bf16_swz_kernel<<<gemmGrid, 256, 0, stream>>>(Abuf, wWv, Pv, Cc, Cc);
    mix_swz_kernel<<<mixGrid, 256, 0, stream>>>(x, tmg, Abuf);
    gemm_bf16_swz_kernel<<<gemmGrid, 256, 0, stream>>>(Abuf, wWg, Pg, Cc, Cc);

    // sequential time scan; final state -> tail of d_out
    scan_kernel<<<Bx * Hh, 64, 0, stream>>>(Pr, Pk, Pv, state, decay, faaaa,
                                            Sout, out + BTC);

    // groupnorm + silu gate -> swizzled bf16 input for output projection
    gnorm_swz_kernel<<<(int)(BT * 4), 256, 0, stream>>>(Sout, Pg, gamma, beta, Xo);

    // output projection -> d_out
    gemm_bf16_swz_kernel<<<gemmGrid, 256, 0, stream>>>(Xo, wWo, out, Cc, Cc);
}
